// GatedLinearAttention_22282290332311
// MI455X (gfx1250) — matrix operand order repaired and checked by reading, not compiled, not run on hardware
//
#include <hip/hip_runtime.h>
#include <hip/hip_bf16.h>

// ---------------------------------------------------------------------------
// Types for CDNA5 WMMA (wave32): bf16 16x16x32, fp32 accumulate
// ---------------------------------------------------------------------------
typedef __attribute__((ext_vector_type(16))) __bf16 bf16x16;
typedef __attribute__((ext_vector_type(8)))  __bf16 bf16x8;
typedef __attribute__((ext_vector_type(8)))  float  v8f;

__device__ __forceinline__ v8f wmma_bf16(bf16x16 a, bf16x16 b, v8f c) {
  return __builtin_amdgcn_wmma_f32_16x16x32_bf16(false, a, false, b,
                                                 (short)0, c, false, false);
}

// A-operand fragment (16x32, M x K), row-major source with leading dim ld.
// lane(0-15)=row r, half=lane>>4; element e(0..7) -> k=half*8+e,
// e(8..15) -> k=16+half*8+(e-8).  => two contiguous 16B reads per lane.
__device__ __forceinline__ bf16x16 load_frag_a(const __bf16* base, int ld) {
  const int lane = threadIdx.x & 31;
  const int half = lane >> 4, r = lane & 15;
  const __bf16* p = base + r * ld + half * 8;
  union { bf16x16 v; bf16x8 h[2]; } u;
  u.h[0] = *(const bf16x8*)(p);
  u.h[1] = *(const bf16x8*)(p + 16);
  return u.v;
}

// B-operand fragment (32x16, K x N), from K-contiguous storage Bt[n][k].
// lane(0-15)=col n, half=lane>>4; e(0..7) -> k=half*8+e, e(8..15) -> k=16+half*8+(e-8).
__device__ __forceinline__ bf16x16 load_frag_b(const __bf16* base, int ld) {
  const int lane = threadIdx.x & 31;
  const int half = lane >> 4, n = lane & 15;
  const __bf16* p = base + n * ld + half * 8;
  union { bf16x16 v; bf16x8 h[2]; } u;
  u.h[0] = *(const bf16x8*)(p);
  u.h[1] = *(const bf16x8*)(p + 16);
  return u.v;
}

// CDNA5 async global->LDS copy (16B per lane), tracked by ASYNCcnt.
__device__ __forceinline__ void async_ld_b128(unsigned int lds, const void* g) {
  asm volatile("global_load_async_to_lds_b128 %0, %1, off"
               :: "v"(lds), "v"(g) : "memory");
}
__device__ __forceinline__ void wait_async0() {
  asm volatile("s_wait_asynccnt 0" ::: "memory");
}

// ---------------------------------------------------------------------------
// Elementwise: fp32 -> bf16 convert
// ---------------------------------------------------------------------------
__global__ __launch_bounds__(256) void cvt_bf16_kernel(const float* __restrict__ in,
                                                       __bf16* __restrict__ out,
                                                       int n) {
  int i = blockIdx.x * 256 + threadIdx.x;
  if (i < n) out[i] = (__bf16)in[i];
}

// ---------------------------------------------------------------------------
// Transpose 2048x2048 fp32 -> bf16  (Wt[n][k] = W[k][n])
// ---------------------------------------------------------------------------
__global__ __launch_bounds__(256) void transpose_bf16_kernel(const float* __restrict__ W,
                                                             __bf16* __restrict__ Wt) {
  __shared__ float tile[32][33];
  const int bx = blockIdx.x;              // n-tile
  const int by = blockIdx.y;              // k-tile
  const int tx = threadIdx.x & 31;
  const int ty = threadIdx.x >> 5;        // 0..7
#pragma unroll
  for (int i = 0; i < 4; ++i)
    tile[ty + i * 8][tx] = W[(size_t)(by * 32 + ty + i * 8) * 2048 + bx * 32 + tx];
  __syncthreads();
#pragma unroll
  for (int i = 0; i < 4; ++i)
    Wt[(size_t)(bx * 32 + ty + i * 8) * 2048 + by * 32 + tx] =
        (__bf16)tile[tx][ty + i * 8];
}

// ---------------------------------------------------------------------------
// gk bottleneck stage 1: h[t][16] = x[t][:] @ Wgk1  (fp32)
// ---------------------------------------------------------------------------
__global__ __launch_bounds__(256) void gk1_kernel(const float* __restrict__ x,
                                                  const float* __restrict__ W1,
                                                  float* __restrict__ hbuf) {
  const int t = blockIdx.x;
  const int tid = threadIdx.x;
  float acc[16];
#pragma unroll
  for (int j = 0; j < 16; ++j) acc[j] = 0.f;
  const float* xp = x + (size_t)t * 2048;
  for (int kk = tid; kk < 2048; kk += 256) {
    float xv = xp[kk];
    const float* wr = W1 + kk * 16;
#pragma unroll
    for (int j = 0; j < 16; ++j) acc[j] += xv * wr[j];
  }
  __shared__ float red[256 * 16];
#pragma unroll
  for (int j = 0; j < 16; ++j) red[tid * 16 + j] = acc[j];
  __syncthreads();
  if (tid < 16) {
    float s = 0.f;
    for (int i = 0; i < 256; ++i) s += red[i * 16 + tid];
    hbuf[t * 16 + tid] = s;
  }
}

// ---------------------------------------------------------------------------
// gk stage 2: gk = log_sigmoid(h @ Wgk2 + b) / 16
// ---------------------------------------------------------------------------
__global__ __launch_bounds__(256) void gk2_kernel(const float* __restrict__ hbuf,
                                                  const float* __restrict__ W2,
                                                  const float* __restrict__ b2,
                                                  float* __restrict__ gkout) {
  const int bid = blockIdx.x;
  const int t = bid >> 3;
  const int d = (bid & 7) * 256 + threadIdx.x;
  __shared__ float hs[16];
  if (threadIdx.x < 16) hs[threadIdx.x] = hbuf[t * 16 + threadIdx.x];
  __syncthreads();
  float a = b2[d];
#pragma unroll
  for (int j = 0; j < 16; ++j) a += hs[j] * W2[j * 2048 + d];
  float ls = fminf(a, 0.f) - log1pf(__expf(-fabsf(a)));
  gkout[(size_t)t * 2048 + d] = ls * (1.f / 16.f);
}

// ---------------------------------------------------------------------------
// Tiled bf16 WMMA GEMM, fixed 8192 x 2048 x 2048.
// C fp32 = A[M,K] (bf16 row-major) x B (Bt[N,K] bf16).
// 128x128 tile, BK=32, 8 waves, async-to-LDS double buffering.
// ---------------------------------------------------------------------------
__global__ __launch_bounds__(256) void gemm_bf16_kernel(const __bf16* __restrict__ A,
                                                        const __bf16* __restrict__ Bt,
                                                        float* __restrict__ C) {
  constexpr int K = 2048;
  constexpr int NK = 64;                    // K / 32
  constexpr int BUF = 128 * 32;             // elements per buffer
  __shared__ __bf16 As[2 * BUF];
  __shared__ __bf16 Bs[2 * BUF];
  const int tid = threadIdx.x;
  const int m0 = blockIdx.y * 128;
  const int n0 = blockIdx.x * 128;
  const int w = tid >> 5;
  const int wm = (w >> 2) * 64;             // 0 or 64   (4 M-tiles)
  const int wn = (w & 3) * 32;              // 0..96     (2 N-tiles)
  const int lane = tid & 31, half = lane >> 4, ln = lane & 15;

  v8f acc[4][2];
#pragma unroll
  for (int i = 0; i < 4; ++i)
#pragma unroll
    for (int j = 0; j < 2; ++j)
#pragma unroll
      for (int e = 0; e < 8; ++e) acc[i][j][e] = 0.f;

  const int rowL = tid >> 1;                // 0..127
  const int colL = (tid & 1) * 16;          // 0 or 16
  const __bf16* Ag = A + (size_t)(m0 + rowL) * K + colL;
  const __bf16* Bg = Bt + (size_t)(n0 + rowL) * K + colL;
  const unsigned int aL =
      (unsigned int)(uintptr_t)(As + rowL * 32 + colL);
  const unsigned int bL =
      (unsigned int)(uintptr_t)(Bs + rowL * 32 + colL);

  // prologue: tile 0 -> buffer 0
  async_ld_b128(aL, Ag);
  async_ld_b128(aL + 16, Ag + 8);
  async_ld_b128(bL, Bg);
  async_ld_b128(bL + 16, Bg + 8);

  for (int it = 0; it < NK; ++it) {
    wait_async0();            // current tile resident (this wave's copies)
    __syncthreads();          // all waves' copies resident; prev compute done
    if (it + 1 < NK) {
      const __bf16* Agn = Ag + (it + 1) * 32;
      const __bf16* Bgn = Bg + (it + 1) * 32;
      const unsigned int off = (unsigned int)(((it + 1) & 1) * (BUF * 2));
      async_ld_b128(aL + off, Agn);
      async_ld_b128(aL + off + 16, Agn + 8);
      async_ld_b128(bL + off, Bgn);
      async_ld_b128(bL + off + 16, Bgn + 8);
    }
    const __bf16* Asb = As + (it & 1) * BUF;
    const __bf16* Bsb = Bs + (it & 1) * BUF;
    bf16x16 fa[4], fb[2];
#pragma unroll
    for (int i = 0; i < 4; ++i) fa[i] = load_frag_a(Asb + (wm + i * 16) * 32, 32);
#pragma unroll
    for (int j = 0; j < 2; ++j) fb[j] = load_frag_b(Bsb + (wn + j * 16) * 32, 32);
#pragma unroll
    for (int i = 0; i < 4; ++i)
#pragma unroll
      for (int j = 0; j < 2; ++j) acc[i][j] = wmma_bf16(fa[i], fb[j], acc[i][j]);
  }

#pragma unroll
  for (int i = 0; i < 4; ++i)
#pragma unroll
    for (int j = 0; j < 2; ++j) {
      float* Cp = C + (size_t)(m0 + wm + i * 16 + 8 * half) * 2048 +
                  (n0 + wn + j * 16 + ln);
#pragma unroll
      for (int e = 0; e < 8; ++e) Cp[e * 2048] = acc[i][j][e];
    }
}

// ---------------------------------------------------------------------------
// Chunked GLA recurrence. One workgroup per (batch, head); 64 sequential
// chunks of 64 tokens. State S (128x128 fp32) lives in WMMA accumulators.
// ---------------------------------------------------------------------------
#define SB_LD 136   // padded leading dim for Sbf (bank-conflict-free dumps)
#define TR_LD 72    // padded leading dim for transposed 128x64 buffers

__global__ __launch_bounds__(256) void gla_chunk_kernel(const float* __restrict__ q,
                                                        const float* __restrict__ k,
                                                        const float* __restrict__ v,
                                                        const float* __restrict__ gk,
                                                        float* __restrict__ o,
                                                        float* __restrict__ Sout) {
  const int bh = blockIdx.x;                 // 0..31
  const int batch = bh >> 4, h = bh & 15;
  const int tid = threadIdx.x;
  const int lane = tid & 31, w = tid >> 5;
  const int half = lane >> 4, ln = lane & 15;

  __shared__ float  Gs[64 * 128];            // 32 KB: per-chunk cumsum of gk
  __shared__ __bf16 Sbf[128 * SB_LD];        // 34 KB: S^T in bf16 (B operand)
  __shared__ __bf16 bufX[128 * TR_LD];       // 18 KB: Qg / Qh / Kw^T
  __shared__ __bf16 bufY[128 * TR_LD];       // 18 KB: Kh / V^T
  __shared__ __bf16 Abf[64 * 64];            //  8 KB: masked A

  // S tiling: 2(M) x 4(N) waves, each 64x32 region = 4x2 tiles
  const int sm0 = (w >> 2) * 64;
  const int sn0 = (w & 3) * 32;
  // o tiling (64x128): each wave 32x32 = 2x2 tiles
  const int om0 = (w >> 2) * 32;
  const int on0 = (w & 3) * 32;
  // A tiling (64x64): each wave 32x16 = 2x1 tiles
  const int am0 = om0;
  const int an0 = (w & 3) * 16;

  v8f S[4][2];
#pragma unroll
  for (int i = 0; i < 4; ++i)
#pragma unroll
    for (int j = 0; j < 2; ++j)
#pragma unroll
      for (int e = 0; e < 8; ++e) S[i][j][e] = 0.f;

  for (int i = tid; i < 128 * SB_LD; i += 256) Sbf[i] = (__bf16)0.f;
  __syncthreads();

  const float scale = 0.08838834764831845f;  // D^-0.5
  const size_t base = (size_t)batch * 4096 * 2048 + (size_t)h * 128;

  for (int ch = 0; ch < 64; ++ch) {
    const size_t cbase = base + (size_t)(ch * 64) * 2048;

    // prefetch next chunk (one 128B line per thread per matrix)
    if (ch + 1 < 64) {
      const size_t nb = cbase + (size_t)64 * 2048 +
                        (size_t)(tid >> 2) * 2048 + (tid & 3) * 32;
      __builtin_prefetch(q + nb, 0, 1);
      __builtin_prefetch(k + nb, 0, 1);
      __builtin_prefetch(v + nb, 0, 1);
      __builtin_prefetch(gk + nb, 0, 1);
    }

    // --- 1) G = cumsum(gk) within chunk -------------------------------
    if (tid < 128) {
      float s = 0.f;
      for (int c = 0; c < 64; ++c) {
        s += gk[cbase + (size_t)c * 2048 + tid];
        Gs[c * 128 + tid] = s;
      }
    }
    __syncthreads();

    // --- 2) Qg = Q * scale * exp(G) -----------------------------------
    for (int i = tid; i < 8192; i += 256) {
      int m = i >> 7, d = i & 127;
      float qv = q[cbase + (size_t)m * 2048 + d];
      bufX[m * 128 + d] = (__bf16)(qv * scale * __expf(Gs[i]));
    }
    __syncthreads();

    // --- 3) o_inter = Qg @ S  (64x128x128) ----------------------------
    v8f oa[2][2];
#pragma unroll
    for (int i = 0; i < 2; ++i)
#pragma unroll
      for (int j = 0; j < 2; ++j)
#pragma unroll
        for (int e = 0; e < 8; ++e) oa[i][j][e] = 0.f;
    for (int kk = 0; kk < 128; kk += 32) {
      bf16x16 fa0 = load_frag_a(bufX + om0 * 128 + kk, 128);
      bf16x16 fa1 = load_frag_a(bufX + (om0 + 16) * 128 + kk, 128);
      bf16x16 fb0 = load_frag_b(Sbf + on0 * SB_LD + kk, SB_LD);
      bf16x16 fb1 = load_frag_b(Sbf + (on0 + 16) * SB_LD + kk, SB_LD);
      oa[0][0] = wmma_bf16(fa0, fb0, oa[0][0]);
      oa[0][1] = wmma_bf16(fa0, fb1, oa[0][1]);
      oa[1][0] = wmma_bf16(fa1, fb0, oa[1][0]);
      oa[1][1] = wmma_bf16(fa1, fb1, oa[1][1]);
    }
    __syncthreads();

    // --- 4) Qh, Kh; A = Qh @ Kh^T; mask tril --------------------------
    for (int i = tid; i < 8192; i += 256) {
      int m = i >> 7, d = i & 127;
      float qv = q[cbase + (size_t)m * 2048 + d];
      float kv = k[cbase + (size_t)m * 2048 + d];
      bufX[m * 128 + d] = (__bf16)(qv * scale * __expf(Gs[i] - Gs[d]));
      bufY[m * 128 + d] = (__bf16)(kv * __expf(Gs[d] - Gs[i]));
    }
    __syncthreads();
    v8f aa[2];
#pragma unroll
    for (int i = 0; i < 2; ++i)
#pragma unroll
      for (int e = 0; e < 8; ++e) aa[i][e] = 0.f;
    for (int kk = 0; kk < 128; kk += 32) {
      bf16x16 fa0 = load_frag_a(bufX + am0 * 128 + kk, 128);
      bf16x16 fa1 = load_frag_a(bufX + (am0 + 16) * 128 + kk, 128);
      bf16x16 fb = load_frag_b(bufY + an0 * 128 + kk, 128);
      aa[0] = wmma_bf16(fa0, fb, aa[0]);
      aa[1] = wmma_bf16(fa1, fb, aa[1]);
    }
    __syncthreads();
#pragma unroll
    for (int i = 0; i < 2; ++i)
#pragma unroll
      for (int e = 0; e < 8; ++e) {
        int m = am0 + i * 16 + e + 8 * half;
        int n = an0 + ln;
        Abf[m * 64 + n] = (__bf16)((n <= m) ? aa[i][e] : 0.f);
      }

    // --- 5) V^T, Kw^T (padded transposed layout) ----------------------
    for (int i = tid; i < 8192; i += 256) {
      int s = i >> 7, d = i & 127;
      float vv = v[cbase + (size_t)s * 2048 + d];
      float kv = k[cbase + (size_t)s * 2048 + d];
      bufY[d * TR_LD + s] = (__bf16)vv;
      bufX[d * TR_LD + s] = (__bf16)(kv * __expf(Gs[63 * 128 + d] - Gs[i]));
    }
    __syncthreads();

    // --- o_intra += A @ V  (64x128x64) --------------------------------
    for (int kk = 0; kk < 64; kk += 32) {
      bf16x16 fa0 = load_frag_a(Abf + om0 * 64 + kk, 64);
      bf16x16 fa1 = load_frag_a(Abf + (om0 + 16) * 64 + kk, 64);
      bf16x16 fb0 = load_frag_b(bufY + on0 * TR_LD + kk, TR_LD);
      bf16x16 fb1 = load_frag_b(bufY + (on0 + 16) * TR_LD + kk, TR_LD);
      oa[0][0] = wmma_bf16(fa0, fb0, oa[0][0]);
      oa[0][1] = wmma_bf16(fa0, fb1, oa[0][1]);
      oa[1][0] = wmma_bf16(fa1, fb0, oa[1][0]);
      oa[1][1] = wmma_bf16(fa1, fb1, oa[1][1]);
    }
#pragma unroll
    for (int i = 0; i < 2; ++i)
#pragma unroll
      for (int j = 0; j < 2; ++j) {
        float* op = o + cbase + (size_t)(om0 + i * 16 + 8 * half) * 2048 +
                    (on0 + j * 16 + ln);
#pragma unroll
        for (int e = 0; e < 8; ++e) op[e * 2048] = oa[i][j][e];
      }

    // --- 6) S = S * exp(gl) + Kw^T @ V  (128x128x64) ------------------
#pragma unroll
    for (int i = 0; i < 4; ++i)
#pragma unroll
      for (int e = 0; e < 8; ++e) {
        float sc = __expf(Gs[63 * 128 + sm0 + i * 16 + e + 8 * half]);
        S[i][0][e] *= sc;
        S[i][1][e] *= sc;
      }
    for (int kk = 0; kk < 64; kk += 32) {
      bf16x16 fa[4], fb[2];
#pragma unroll
      for (int i = 0; i < 4; ++i)
        fa[i] = load_frag_a(bufX + (sm0 + i * 16) * TR_LD + kk, TR_LD);
#pragma unroll
      for (int j = 0; j < 2; ++j)
        fb[j] = load_frag_b(bufY + (sn0 + j * 16) * TR_LD + kk, TR_LD);
#pragma unroll
      for (int i = 0; i < 4; ++i)
#pragma unroll
        for (int j = 0; j < 2; ++j) S[i][j] = wmma_bf16(fa[i], fb[j], S[i][j]);
    }
    __syncthreads();

    // --- 7) Sbf[n][m] = bf16(S[m][n]) for next chunk's o_inter --------
#pragma unroll
    for (int i = 0; i < 4; ++i)
#pragma unroll
      for (int j = 0; j < 2; ++j)
#pragma unroll
        for (int e = 0; e < 8; ++e) {
          int m = sm0 + i * 16 + e + 8 * half;
          int n = sn0 + j * 16 + ln;
          Sbf[n * SB_LD + m] = (__bf16)S[i][j][e];
        }
    __syncthreads();
  }

  // S_final -> output tail
  const size_t so = (size_t)bh * 16384;
#pragma unroll
  for (int i = 0; i < 4; ++i)
#pragma unroll
    for (int j = 0; j < 2; ++j) {
      float* sp = Sout + so + (size_t)(sm0 + i * 16 + 8 * half) * 128 +
                  (sn0 + j * 16 + ln);
#pragma unroll
      for (int e = 0; e < 8; ++e) sp[e * 128] = S[i][j][e];
    }
}

// ---------------------------------------------------------------------------
// RMS-norm over D=128 per (token, head) * gnorm_w, times silu(g) -> bf16
// ---------------------------------------------------------------------------
__global__ __launch_bounds__(128) void norm_gate_kernel(const float* __restrict__ o,
                                                        const float* __restrict__ g,
                                                        const float* __restrict__ gw,
                                                        __bf16* __restrict__ y) {
  const int row = blockIdx.x;              // token*16 + h
  const int d = threadIdx.x;
  const int token = row >> 4, h = row & 15;
  const size_t idx = (size_t)token * 2048 + h * 128 + d;
  float val = o[idx];
  __shared__ float red[128];
  red[d] = val * val;
  __syncthreads();
  for (int s = 64; s > 0; s >>= 1) {
    if (d < s) red[d] += red[d + s];
    __syncthreads();
  }
  float r = rsqrtf(red[0] * (1.f / 128.f) + 1e-5f);
  float gv = g[idx];
  float silu = gv / (1.f + __expf(-gv));
  y[idx] = (__bf16)(val * r * gw[d] * silu);
}

// ---------------------------------------------------------------------------
// Host launcher
// ---------------------------------------------------------------------------
extern "C" void kernel_launch(void* const* d_in, const int* in_sizes, int n_in,
                              void* d_out, int out_size, void* d_ws, size_t ws_size,
                              hipStream_t stream) {
  const float* x    = (const float*)d_in[0];
  const float* Wq   = (const float*)d_in[1];
  const float* Wk   = (const float*)d_in[2];
  const float* Wv   = (const float*)d_in[3];
  const float* Wgk1 = (const float*)d_in[4];
  const float* Wgk2 = (const float*)d_in[5];
  const float* bgk2 = (const float*)d_in[6];
  const float* Wg   = (const float*)d_in[7];
  const float* gnw  = (const float*)d_in[8];
  const float* Wo   = (const float*)d_in[9];

  const int M = 8192, N = 2048, K = 2048;
  const size_t MN = (size_t)M * N;          // 16,777,216

  // workspace carve-up
  char* ws = (char*)d_ws;
  __bf16* x_bf  = (__bf16*)ws;                          ws += MN * 2;
  __bf16* Wt_q  = (__bf16*)ws;                          ws += (size_t)N * K * 2;
  __bf16* Wt_k  = (__bf16*)ws;                          ws += (size_t)N * K * 2;
  __bf16* Wt_v  = (__bf16*)ws;                          ws += (size_t)N * K * 2;
  __bf16* Wt_g  = (__bf16*)ws;                          ws += (size_t)N * K * 2;
  __bf16* Wt_o  = (__bf16*)ws;                          ws += (size_t)N * K * 2;
  float*  qb    = (float*)ws;                           ws += MN * 4;
  float*  kb    = (float*)ws;                           ws += MN * 4;
  float*  vb    = (float*)ws;                           ws += MN * 4;
  float*  gkb   = (float*)ws;                           ws += MN * 4;
  float*  gb    = (float*)ws;                           ws += MN * 4;
  float*  ob    = (float*)ws;                           ws += MN * 4;
  float*  hb    = (float*)ws;                           ws += (size_t)M * 16 * 4;
  __bf16* yb    = (__bf16*)ws;                          ws += MN * 2;

  float* out_main = (float*)d_out;
  float* out_S    = out_main + MN;

  // 1) conversions
  cvt_bf16_kernel<<<(int)(MN / 256), 256, 0, stream>>>(x, x_bf, (int)MN);
  dim3 tgrid(64, 64);
  transpose_bf16_kernel<<<tgrid, 256, 0, stream>>>(Wq, Wt_q);
  transpose_bf16_kernel<<<tgrid, 256, 0, stream>>>(Wk, Wt_k);
  transpose_bf16_kernel<<<tgrid, 256, 0, stream>>>(Wv, Wt_v);
  transpose_bf16_kernel<<<tgrid, 256, 0, stream>>>(Wg, Wt_g);
  transpose_bf16_kernel<<<tgrid, 256, 0, stream>>>(Wo, Wt_o);

  // 2) gk bottleneck (fp32)
  gk1_kernel<<<M, 256, 0, stream>>>(x, Wgk1, hb);
  gk2_kernel<<<M * 8, 256, 0, stream>>>(hb, Wgk2, bgk2, gkb);

  // 3) projections via WMMA GEMM (async double-buffered)
  dim3 ggrid(N / 128, M / 128);
  gemm_bf16_kernel<<<ggrid, 256, 0, stream>>>(x_bf, Wt_q, qb);
  gemm_bf16_kernel<<<ggrid, 256, 0, stream>>>(x_bf, Wt_k, kb);
  gemm_bf16_kernel<<<ggrid, 256, 0, stream>>>(x_bf, Wt_v, vb);
  gemm_bf16_kernel<<<ggrid, 256, 0, stream>>>(x_bf, Wt_g, gb);

  // 4) chunked recurrence (writes o and S_final)
  gla_chunk_kernel<<<32, 256, 0, stream>>>(qb, kb, vb, gkb, ob, out_S);

  // 5) norm + gate, then output projection
  norm_gate_kernel<<<M * 16, 128, 0, stream>>>(ob, gb, gnw, yb);
  gemm_bf16_kernel<<<ggrid, 256, 0, stream>>>(yb, Wt_o, out_main);
}
